// ASLGraphClassifier_40389872451908
// MI455X (gfx1250) — compile-verified, matrix operand order checked by reading, not run on hardware
//
#include <hip/hip_runtime.h>
#include <hip/hip_bf16.h>
#include <math.h>

// ---- problem constants (match reference) ----
#define N_NODES  32768
#define N_EDGES  262144
#define F_IN     128
#define H1_DIM   512
#define H2_DIM   1024
#define G_GRAPHS 1024
#define NPG      32
#define N_CLS    20
#define EPS_BN   1e-5f
#define EPS_LN   1e-5f

typedef __attribute__((ext_vector_type(16))) __bf16 v16bf;
typedef __attribute__((ext_vector_type(8)))  float  v8f;

// =====================================================================
// Block-tiled WMMA GEMM: C[M,Nc] = A[M,K] @ B[K,Nc]; f32 in/out, bf16
// multiply, f32 accumulate (V_WMMA_F32_16X16X32_BF16).
//
// Workgroup = 256 threads = 8 waves arranged 4(M) x 2(N).
// Block tile = 128(M) x 64(N); wave tile = 32x32 = 2x2 WMMA tiles.
// Per K-step(32): stage A(128x32) + B(32x64) to LDS as bf16 with
// coalesced loads, then 4 WMMAs per wave from LDS fragments.
//
// Fragment layouts per ISA 7.12.2 (wave32):
//   A 16x32 bf16 : lane L -> row m=L&15, half=L>>4;
//                  elem e<8 -> K=half*8+e ; e>=8 -> K=16+half*8+(e-8)
//   B 32x16 bf16 : lane L -> col n=L&15, same K mapping
//   C/D 16x16 f32: VGPR r -> row = (L>>4)*8 + r, col = L&15
// =====================================================================
__global__ __launch_bounds__(256) void k_gemm_bf16(
    const float* __restrict__ A, const float* __restrict__ B,
    float* __restrict__ C, int M, int K, int Nc) {
  __shared__ __bf16 As[128][34];   // +2 pad vs 32 to de-bank
  __shared__ __bf16 Bs[32][66];    // +2 pad vs 64

  const int tid   = threadIdx.x;
  const int lane  = tid & 31;
  const int wv    = tid >> 5;      // 0..7
  const int waveM = wv & 3;        // 4 waves along M
  const int waveN = wv >> 2;       // 2 waves along N
  const int l15   = lane & 15;
  const int hl    = lane >> 4;

  const int blocksN = Nc >> 6;                       // Nc / 64
  const int bm = (blockIdx.x / blocksN) << 7;        // * 128
  const int bn = (blockIdx.x % blocksN) << 6;        // * 64

  // staging assignments (coalesced along K for A, along N for B)
  const int arow = tid >> 1;            // 0..127
  const int akh  = (tid & 1) << 4;      // 0 or 16
  const int brow = tid >> 3;            // 0..31
  const int bcol = (tid & 7) << 3;      // 0..56

  v8f acc[2][2] = {};

  for (int kb = 0; kb < K; kb += 32) {
    __syncthreads();                    // protect LDS from prior-iter readers
    {
      const float* pa = A + (size_t)(bm + arow) * K + kb + akh;
#pragma unroll
      for (int e = 0; e < 16; ++e) As[arow][akh + e] = (__bf16)pa[e];
    }
    {
      const float* pb = B + (size_t)(kb + brow) * Nc + bn + bcol;
#pragma unroll
      for (int e = 0; e < 8; ++e) Bs[brow][bcol + e] = (__bf16)pb[e];
    }
    __syncthreads();

    v16bf af[2], bg[2];
#pragma unroll
    for (int t = 0; t < 2; ++t) {
      const int r = (waveM << 5) + (t << 4) + l15;
      const int c = (waveN << 5) + (t << 4) + l15;
#pragma unroll
      for (int e = 0; e < 8; ++e) {
        af[t][e]     = As[r][(hl << 3) + e];
        af[t][e + 8] = As[r][16 + (hl << 3) + e];
        bg[t][e]     = Bs[(hl << 3) + e][c];
        bg[t][e + 8] = Bs[16 + (hl << 3) + e][c];
      }
    }
#pragma unroll
    for (int tm = 0; tm < 2; ++tm)
#pragma unroll
      for (int tn = 0; tn < 2; ++tn)
        acc[tm][tn] = __builtin_amdgcn_wmma_f32_16x16x32_bf16(
            false, af[tm], false, bg[tn], (short)0, acc[tm][tn], false, false);
  }

#pragma unroll
  for (int tm = 0; tm < 2; ++tm) {
    const int rb = bm + (waveM << 5) + (tm << 4) + (hl << 3);
#pragma unroll
    for (int tn = 0; tn < 2; ++tn) {
      const int cc = bn + (waveN << 5) + (tn << 4) + l15;
#pragma unroll
      for (int r = 0; r < 8; ++r)
        C[(size_t)(rb + r) * Nc + cc] = acc[tm][tn][r];
    }
  }
}

// ---------------- degree / normalization ----------------
__global__ void k_fill1(float* p, int n) {
  int i = blockIdx.x * blockDim.x + threadIdx.x;
  if (i < n) p[i] = 1.0f;                    // self-loop contribution
}
__global__ void k_degcount(const int* __restrict__ dst, float* __restrict__ deg, int e) {
  int i = blockIdx.x * blockDim.x + threadIdx.x;
  if (i < e) atomicAdd(&deg[dst[i]], 1.0f);
}
__global__ void k_rsqrt(const float* __restrict__ deg, float* __restrict__ dis, int n) {
  int i = blockIdx.x * blockDim.x + threadIdx.x;
  if (i < n) dis[i] = rsqrtf(deg[i]);        // deg >= 1 always (self-loops)
}

// agg[i][c] = dis[i]^2 * xw[i][c]   (self-loop term initializes the buffer)
__global__ void k_selfloop(const float* __restrict__ dis, const float* __restrict__ xw,
                           float* __restrict__ agg, int total, int H) {
  int i = blockIdx.x * blockDim.x + threadIdx.x;
  if (i >= total) return;
  float w = dis[i / H];
  agg[i] = w * w * xw[i];
}

// per edge: agg[dst] += dis[s]*dis[d] * xw[src]   (chunked over channels)
__global__ void k_edge_scatter(const int* __restrict__ src, const int* __restrict__ dst,
                               const float* __restrict__ dis, const float* __restrict__ xw,
                               float* __restrict__ agg, int H, int chunks) {
  const int e = blockIdx.x / chunks;
  const int c = (blockIdx.x % chunks) * blockDim.x + threadIdx.x;
  const int s = src[e], d = dst[e];
  const float w = dis[s] * dis[d];
  atomicAdd(&agg[(size_t)d * H + c], w * xw[(size_t)s * H + c]);
}

// per-column sum/sumsq partials (coalesced across channels), atomic combine
__global__ void k_colstats(const float* __restrict__ x, float* __restrict__ sum,
                           float* __restrict__ sumsq, int rows, int H, int rowsPerBlk) {
  const int c  = blockIdx.x * blockDim.x + threadIdx.x;
  const int r0 = blockIdx.y * rowsPerBlk;
  float s = 0.f, ss = 0.f;
  for (int r = r0; r < r0 + rowsPerBlk; ++r) {
    float v = x[(size_t)r * H + c];
    s += v; ss += v * v;
  }
  atomicAdd(&sum[c], s);
  atomicAdd(&sumsq[c], ss);
}

// in-place BatchNorm(training stats, biased var) + ReLU
__global__ void k_bn_relu(float* __restrict__ x, const float* __restrict__ sum,
                          const float* __restrict__ sumsq, const float* __restrict__ w,
                          const float* __restrict__ b, int rows, int H) {
  int i = blockIdx.x * blockDim.x + threadIdx.x;
  if (i >= rows * H) return;
  int c = i % H;
  float inv = 1.0f / (float)rows;
  float mu  = sum[c] * inv;
  float var = sumsq[c] * inv - mu * mu;
  float v = (x[i] - mu) * rsqrtf(var + EPS_BN) * w[c] + b[c];
  x[i] = v > 0.f ? v : 0.f;
}

// global (whole-tensor) sum / sumsq with f64 atomics
__global__ void k_gstats(const float* __restrict__ x, int total, double* __restrict__ acc) {
  float s = 0.f, ss = 0.f;
  for (int i = blockIdx.x * blockDim.x + threadIdx.x; i < total;
       i += gridDim.x * blockDim.x) {
    float v = x[i];
    s += v; ss += v * v;
  }
  __shared__ float sh[256], sh2[256];
  sh[threadIdx.x] = s; sh2[threadIdx.x] = ss;
  __syncthreads();
  for (int st = 128; st > 0; st >>= 1) {
    if ((int)threadIdx.x < st) { sh[threadIdx.x] += sh[threadIdx.x + st];
                                 sh2[threadIdx.x] += sh2[threadIdx.x + st]; }
    __syncthreads();
  }
  if (threadIdx.x == 0) {
    atomicAdd(&acc[0], (double)sh[0]);
    atomicAdd(&acc[1], (double)sh2[0]);
  }
}

// fused graph-LayerNorm apply + segment_max over 32 contiguous nodes/graph.
// LN affine is per-channel monotone: max for w>=0, min for w<0.
__global__ void k_ln_pool(const float* __restrict__ h, const double* __restrict__ gs,
                          const float* __restrict__ w, const float* __restrict__ b,
                          float* __restrict__ pooled, int H, int total) {
  int idx = blockIdx.x * blockDim.x + threadIdx.x;     // g*H + c
  int g = idx / H, c = idx % H;
  double mu  = gs[0] / (double)total;
  double var = gs[1] / (double)total - mu * mu;
  float stdv = sqrtf(var > 0.0 ? (float)var : 0.f);
  float inv  = 1.0f / (stdv + EPS_LN);
  const float* p = h + (size_t)g * NPG * H + c;
  float wc = w[c], m;
  if (wc >= 0.f) {
    m = -3.402823466e38f;
    for (int i = 0; i < NPG; ++i) { float v = p[(size_t)i * H]; m = v > m ? v : m; }
  } else {
    m = 3.402823466e38f;
    for (int i = 0; i < NPG; ++i) { float v = p[(size_t)i * H]; m = v < m ? v : m; }
  }
  pooled[idx] = (m - (float)mu) * inv * wc + b[c];
}

__global__ void k_bias_relu(float* __restrict__ x, const float* __restrict__ b,
                            int total, int H) {
  int i = blockIdx.x * blockDim.x + threadIdx.x;
  if (i >= total) return;
  float v = x[i] + b[i % H];
  x[i] = v > 0.f ? v : 0.f;
}

// head: LN2 apply + lin2 (512x20) + log_softmax; one block per graph
__global__ void k_head(const float* __restrict__ z, const double* __restrict__ gs,
                       const float* __restrict__ lnw, const float* __restrict__ lnb,
                       const float* __restrict__ w2, const float* __restrict__ b2,
                       float* __restrict__ out, int total) {
  __shared__ float zn[H1_DIM];
  __shared__ float logit[N_CLS];
  __shared__ float mx, lse;
  const int g = blockIdx.x;
  double mu  = gs[0] / (double)total;
  double var = gs[1] / (double)total - mu * mu;
  float stdv = sqrtf(var > 0.0 ? (float)var : 0.f);
  float inv  = 1.0f / (stdv + EPS_LN);
  for (int c = threadIdx.x; c < H1_DIM; c += blockDim.x)
    zn[c] = (z[(size_t)g * H1_DIM + c] - (float)mu) * inv * lnw[c] + lnb[c];
  __syncthreads();
  if ((int)threadIdx.x < N_CLS) {
    float a = b2[threadIdx.x];
    for (int k = 0; k < H1_DIM; ++k) a += zn[k] * w2[(size_t)k * N_CLS + threadIdx.x];
    logit[threadIdx.x] = a;
  }
  __syncthreads();
  if (threadIdx.x == 0) {
    float m = -3.402823466e38f;
    for (int c = 0; c < N_CLS; ++c) m = logit[c] > m ? logit[c] : m;
    float s = 0.f;
    for (int c = 0; c < N_CLS; ++c) s += expf(logit[c] - m);
    mx = m; lse = logf(s);
  }
  __syncthreads();
  if ((int)threadIdx.x < N_CLS)
    out[(size_t)g * N_CLS + threadIdx.x] = logit[threadIdx.x] - mx - lse;
}

// =====================================================================
extern "C" void kernel_launch(void* const* d_in, const int* in_sizes, int n_in,
                              void* d_out, int out_size, void* d_ws, size_t ws_size,
                              hipStream_t stream) {
  (void)in_sizes; (void)n_in; (void)out_size; (void)ws_size;
  const float* x       = (const float*)d_in[0];
  const int*   ei      = (const int*)d_in[1];     // [2, E]
  /* d_in[2] = batch: implied by node index (i / 32) */
  const float* conv1_w = (const float*)d_in[3];   // conv biases cancel through BN
  const float* bn1_w   = (const float*)d_in[5];
  const float* bn1_b   = (const float*)d_in[6];
  const float* conv2_w = (const float*)d_in[7];
  const float* bn2_w   = (const float*)d_in[9];
  const float* bn2_b   = (const float*)d_in[10];
  const float* ln1_w   = (const float*)d_in[11];
  const float* ln1_b   = (const float*)d_in[12];
  const float* lin1_w  = (const float*)d_in[13];
  const float* lin1_b  = (const float*)d_in[14];
  const float* ln2_w   = (const float*)d_in[15];
  const float* ln2_b   = (const float*)d_in[16];
  const float* lin2_w  = (const float*)d_in[17];
  const float* lin2_b  = (const float*)d_in[18];
  float* out = (float*)d_out;

  const int* src = ei;
  const int* dst = ei + N_EDGES;

  // ---- workspace layout (ping-pong 128MiB regions + misc) ----
  const size_t MB = 1024 * 1024;
  char* base = (char*)d_ws;
  float* bufA = (float*)base;                 // xw1 -> xw2 -> pooled/z
  float* bufB = (float*)(base + 128 * MB);    // agg1/h1 -> agg2/h2
  char*  misc = base + 256 * MB;
  float*  deg      = (float*)misc;                         // N floats
  float*  dis      = (float*)(misc + 131072);              // N floats
  float*  colsum   = (float*)(misc + 262144);              // H2 floats
  float*  colsumsq = (float*)(misc + 262144 + 4096);       // H2 floats
  double* gacc     = (double*)(misc + 262144 + 8192);      // 2 doubles
  double* gacc2    = gacc + 2;                             // 2 doubles
  float*  pooled   = bufA;                                 // G*H2 (4 MiB)
  float*  zbuf     = (float*)(base + 8 * MB);              // G*H1 (2 MiB)

  const dim3 T(256);

  // ---- degrees (with self-loops) -> dis = deg^-1/2 ----
  k_fill1   <<<N_NODES / 256, T, 0, stream>>>(deg, N_NODES);
  k_degcount<<<N_EDGES / 256, T, 0, stream>>>(dst, deg, N_EDGES);
  k_rsqrt   <<<N_NODES / 256, T, 0, stream>>>(deg, dis, N_NODES);

  // ---- layer 1: xw1 = x @ W1 ----
  k_gemm_bf16<<<(N_NODES / 128) * (H1_DIM / 64), T, 0, stream>>>(
      x, conv1_w, bufA, N_NODES, F_IN, H1_DIM);
  k_selfloop<<<(N_NODES * H1_DIM) / 256, T, 0, stream>>>(dis, bufA, bufB, N_NODES * H1_DIM, H1_DIM);
  k_edge_scatter<<<N_EDGES * (H1_DIM / 256), T, 0, stream>>>(src, dst, dis, bufA, bufB,
                                                             H1_DIM, H1_DIM / 256);
  hipMemsetAsync(colsum, 0, 2 * H2_DIM * sizeof(float), stream);
  k_colstats<<<dim3(H1_DIM / 256, 64), T, 0, stream>>>(bufB, colsum, colsumsq,
                                                       N_NODES, H1_DIM, N_NODES / 64);
  k_bn_relu<<<(N_NODES * H1_DIM) / 256, T, 0, stream>>>(bufB, colsum, colsumsq,
                                                        bn1_w, bn1_b, N_NODES, H1_DIM);

  // ---- layer 2: xw2 = h1 @ W2 ----
  k_gemm_bf16<<<(N_NODES / 128) * (H2_DIM / 64), T, 0, stream>>>(
      bufB, conv2_w, bufA, N_NODES, H1_DIM, H2_DIM);
  k_selfloop<<<(N_NODES * H2_DIM) / 256, T, 0, stream>>>(dis, bufA, bufB, N_NODES * H2_DIM, H2_DIM);
  k_edge_scatter<<<N_EDGES * (H2_DIM / 256), T, 0, stream>>>(src, dst, dis, bufA, bufB,
                                                             H2_DIM, H2_DIM / 256);
  hipMemsetAsync(colsum, 0, 2 * H2_DIM * sizeof(float), stream);
  k_colstats<<<dim3(H2_DIM / 256, 64), T, 0, stream>>>(bufB, colsum, colsumsq,
                                                       N_NODES, H2_DIM, N_NODES / 64);
  k_bn_relu<<<(N_NODES * H2_DIM) / 256, T, 0, stream>>>(bufB, colsum, colsumsq,
                                                        bn2_w, bn2_b, N_NODES, H2_DIM);

  // ---- graph LayerNorm 1 (whole tensor) + fused max pool ----
  hipMemsetAsync(gacc, 0, 4 * sizeof(double), stream);
  k_gstats<<<4096, T, 0, stream>>>(bufB, N_NODES * H2_DIM, gacc);
  k_ln_pool<<<(G_GRAPHS * H2_DIM) / 256, T, 0, stream>>>(bufB, gacc, ln1_w, ln1_b,
                                                         pooled, H2_DIM, N_NODES * H2_DIM);

  // ---- lin1 + ReLU ----
  k_gemm_bf16<<<(G_GRAPHS / 128) * (H1_DIM / 64), T, 0, stream>>>(
      pooled, lin1_w, zbuf, G_GRAPHS, H2_DIM, H1_DIM);
  k_bias_relu<<<(G_GRAPHS * H1_DIM) / 256, T, 0, stream>>>(zbuf, lin1_b,
                                                           G_GRAPHS * H1_DIM, H1_DIM);

  // ---- graph LayerNorm 2 + lin2 + log_softmax ----
  k_gstats<<<1024, T, 0, stream>>>(zbuf, G_GRAPHS * H1_DIM, gacc2);
  k_head<<<G_GRAPHS, T, 0, stream>>>(zbuf, gacc2, ln2_w, ln2_b, lin2_w, lin2_b,
                                     out, G_GRAPHS * H1_DIM);
}
